// GatedRNN_3968549782190
// MI455X (gfx1250) — matrix operand order repaired and checked by reading, not compiled, not run on hardware
//
#include <hip/hip_runtime.h>

typedef __attribute__((ext_vector_type(16))) __bf16 v16bf;
typedef __attribute__((ext_vector_type(8)))  float  v8f;
typedef __attribute__((ext_vector_type(4)))  float  v4f;

#define Bz 64
#define Tz 2048
#define DIMz 256
#define VOCABz 32000
#define MTOT (Bz * Tz)
#define SEG 4
#define NTILES_HEAD (VOCABz / 16)   // 2000

#define WMMA_BF16(A, Bm, C) \
  __builtin_amdgcn_wmma_f32_16x16x32_bf16(false, (A), false, (Bm), (short)0, (C), false, false)

// Split two contiguous v4f's worth of f32 into bf16 hi/lo halves of a fragment.
__device__ __forceinline__ void cvt8(const v4f a, const v4f b, int off,
                                     v16bf& hi, v16bf& lo) {
#pragma unroll
  for (int j = 0; j < 4; ++j) {
    float v = a[j]; __bf16 h = (__bf16)v;
    hi[off + j] = h; lo[off + j] = (__bf16)(v - (float)h);
  }
#pragma unroll
  for (int j = 0; j < 4; ++j) {
    float v = b[j]; __bf16 h = (__bf16)v;
    hi[off + 4 + j] = h; lo[off + 4 + j] = (__bf16)(v - (float)h);
  }
}

// ---------------------------------------------------------------------------
// Pack W_w [256,256] (row-major [N][K]) into bf16 hi/lo B-fragments:
// frag index t = (kc*16 + nt)*32 + lane ; 16 contiguous bf16 per lane.
// ---------------------------------------------------------------------------
__global__ __launch_bounds__(256)
void k_pack_w(const float* __restrict__ Ww,
              __bf16* __restrict__ Whp, __bf16* __restrict__ Wlp) {
  const int t    = blockIdx.x * 256 + threadIdx.x;  // 0..4095
  const int lane = t & 31;
  const int fi   = t >> 5;                          // kc*16 + nt
  const int kc   = fi >> 4;
  const int nt   = fi & 15;
  const int N    = nt * 16 + (lane & 15);
  const int K0   = kc * 32 + (lane >> 4) * 8;
  const float* src = Ww + (size_t)N * DIMz + K0;
  v16bf hi, lo;
  cvt8(*(const v4f*)(src),     *(const v4f*)(src + 4),  0, hi, lo);
  cvt8(*(const v4f*)(src + 16), *(const v4f*)(src + 20), 8, hi, lo);
  *(v16bf*)(Whp + (size_t)t * 16) = hi;
  *(v16bf*)(Wlp + (size_t)t * 16) = lo;
}

// Same packing for head_w [32000,256]: t = (nt*8 + kc)*32 + lane.
__global__ __launch_bounds__(256)
void k_pack_hw(const float* __restrict__ Hw,
               __bf16* __restrict__ Hhp, __bf16* __restrict__ Hlp) {
  const int t    = blockIdx.x * 256 + threadIdx.x;  // 0..511999
  const int lane = t & 31;
  const int fi   = t >> 5;                          // nt*8 + kc
  const int kc   = fi & 7;
  const int nt   = fi >> 3;
  const int N    = nt * 16 + (lane & 15);
  const int K0   = kc * 32 + (lane >> 4) * 8;
  const float* src = Hw + (size_t)N * DIMz + K0;
  v16bf hi, lo;
  cvt8(*(const v4f*)(src),     *(const v4f*)(src + 4),  0, hi, lo);
  cvt8(*(const v4f*)(src + 16), *(const v4f*)(src + 20), 8, hi, lo);
  *(v16bf*)(Hhp + (size_t)t * 16) = hi;
  *(v16bf*)(Hlp + (size_t)t * 16) = lo;
}

// ---------------------------------------------------------------------------
// Kernel 1: block = 128 rows (8 strips).  Phase A: each wave gathers its
// strip's embedding rows, computes the gate dot, converts to bf16 hi/lo A
// fragments stored in LDS.  Phase B: each wave owns 2 N-tiles; per
// (nt,kc) it loads one pre-packed B fragment and sweeps all 8 strips from
// LDS, so B is read once per 128 rows instead of once per 16.
// ---------------------------------------------------------------------------
__global__ __launch_bounds__(256)
void k_gate_cand(const int* __restrict__ x, const float* __restrict__ emb,
                 const __bf16* __restrict__ Whp, const __bf16* __restrict__ Wlp,
                 const float* __restrict__ Wb, const float* __restrict__ Wg,
                 const float* __restrict__ Wgb,
                 float* __restrict__ g_out, float* __restrict__ c_out) {
  extern __shared__ __bf16 smem[];   // [hi|lo], each 8 strips x 8 kc x 32 lanes x 16
  const int lane  = threadIdx.x & 31;
  const int wave  = threadIdx.x >> 5;
  const int half  = lane >> 4;
  const int r15   = lane & 15;
  const int mBase = blockIdx.x * 128;

  // ---- Phase A: stage A fragments for strip == wave ----
  {
    const int rowA = mBase + wave * 16 + r15;
    const float* erow = emb + (size_t)x[rowA] * DIMz;
    float gdot = 0.f;
#pragma unroll
    for (int kc = 0; kc < 8; ++kc) {
      const int kb = kc * 32 + half * 8;
      v4f e0 = *(const v4f*)(erow + kb);
      v4f e1 = *(const v4f*)(erow + kb + 4);
      v4f e2 = *(const v4f*)(erow + kb + 16);
      v4f e3 = *(const v4f*)(erow + kb + 20);
      v4f w0 = *(const v4f*)(Wg + kb);
      v4f w1 = *(const v4f*)(Wg + kb + 4);
      v4f w2 = *(const v4f*)(Wg + kb + 16);
      v4f w3 = *(const v4f*)(Wg + kb + 20);
#pragma unroll
      for (int j = 0; j < 4; ++j)
        gdot += e0[j] * w0[j] + e1[j] * w1[j] + e2[j] * w2[j] + e3[j] * w3[j];
      v16bf hi, lo;
      cvt8(e0, e1, 0, hi, lo);
      cvt8(e2, e3, 8, hi, lo);
      const size_t off = ((size_t)(wave * 8 + kc) * 32 + lane) * 16;
      *(v16bf*)(&smem[off])         = hi;
      *(v16bf*)(&smem[32768 + off]) = lo;
    }
    gdot += __shfl_xor(gdot, 16, 32);
    if (half == 0) {
      float z = gdot + Wgb[0];
      g_out[rowA] = 1.f / (1.f + __expf(-z));
    }
  }
  __syncthreads();

  // ---- Phase B: wave handles N-tiles {2*wave, 2*wave+1} over all 8 strips ----
#pragma unroll 1
  for (int nn = 0; nn < 2; ++nn) {
    const int nt = wave * 2 + nn;
    const int N  = nt * 16 + r15;
    v8f acc[8];
    const v8f vz = {0.f, 0.f, 0.f, 0.f, 0.f, 0.f, 0.f, 0.f};
#pragma unroll
    for (int s = 0; s < 8; ++s) acc[s] = vz;

#pragma unroll 1
    for (int kc = 0; kc < 8; ++kc) {
      const size_t bo = ((size_t)(kc * 16 + nt) * 32 + lane) * 16;
      const v16bf Bh = *(const v16bf*)(Whp + bo);
      const v16bf Bl = *(const v16bf*)(Wlp + bo);
#pragma unroll
      for (int s = 0; s < 8; ++s) {
        const size_t ao = ((size_t)(s * 8 + kc) * 32 + lane) * 16;
        v16bf Ah = *(const v16bf*)(&smem[ao]);
        v16bf Al = *(const v16bf*)(&smem[32768 + ao]);
        acc[s] = WMMA_BF16(Ah, Bh, acc[s]);
        acc[s] = WMMA_BF16(Al, Bh, acc[s]);
        acc[s] = WMMA_BF16(Ah, Bl, acc[s]);
      }
    }
    const float bias = Wb[N];
#pragma unroll
    for (int s = 0; s < 8; ++s) {
#pragma unroll
      for (int rr = 0; rr < 8; ++rr) {
        const int mrow = mBase + s * 16 + half * 8 + rr;
        c_out[(size_t)mrow * DIMz + N] = tanhf(acc[s][rr] + bias);
      }
    }
  }
}

// ---------------------------------------------------------------------------
// Kernel 2a: segmented linear scan (linear recurrence => segment = (S, P)).
// ---------------------------------------------------------------------------
__global__ __launch_bounds__(DIMz)
void k_scan_seg(const float* __restrict__ g, const float* __restrict__ c,
                float* __restrict__ S, float* __restrict__ P) {
  const int seg = blockIdx.x & (SEG - 1);
  const int b   = blockIdx.x >> 2;
  const int d   = threadIdx.x;
  const int len = Tz / SEG;
  const int t0  = seg * len;
  const float* gb = g + (size_t)b * Tz + t0;
  const float* cb = c + ((size_t)b * Tz + t0) * DIMz + d;
  float h = 0.f, p = 1.f;
  for (int t = 0; t < len; ++t) {
    if (t + 16 < len)
      __builtin_prefetch(cb + (size_t)(t + 16) * DIMz, 0, 0);
    float gt = gb[t];
    float ct = cb[(size_t)t * DIMz];
    h = fmaf(gt, h - ct, ct);   // c + g*(h - c)
    p *= gt;
  }
  S[((size_t)seg * Bz + b) * DIMz + d] = h;
  if (d == 0) P[seg * Bz + b] = p;
}

__global__ __launch_bounds__(DIMz)
void k_combine(const float* __restrict__ S, const float* __restrict__ P,
               float* __restrict__ hT) {
  const int b = blockIdx.x, d = threadIdx.x;
  float h = 0.f;
#pragma unroll
  for (int s = 0; s < SEG; ++s)
    h = h * P[s * Bz + b] + S[((size_t)s * Bz + b) * DIMz + d];
  hT[b * DIMz + d] = h;
}

// ---------------------------------------------------------------------------
// Kernel 3: out = hT @ head_w^T + head_b.  Block stages ALL 64 hT rows as
// bf16 hi/lo A fragments in LDS (64 KB); each wave owns 2 N-tiles and sweeps
// all 4 M-strips per pre-packed B fragment, so head_w is read exactly once
// across the grid (33 MB instead of 131 MB) and each B load feeds 12 WMMAs.
// ---------------------------------------------------------------------------
__global__ __launch_bounds__(256)
void k_head(const float* __restrict__ hT,
            const __bf16* __restrict__ Hhp, const __bf16* __restrict__ Hlp,
            const float* __restrict__ Hb, float* __restrict__ out) {
  extern __shared__ __bf16 smem[];   // [hi|lo], each 4 strips x 8 kc x 32 lanes x 16
  const int lane = threadIdx.x & 31;
  const int wave = threadIdx.x >> 5;
  const int half = lane >> 4;
  const int r15  = lane & 15;

  // ---- Phase A: 8 waves cooperatively stage 32 (strip,kc) fragment sets ----
#pragma unroll
  for (int i = 0; i < 4; ++i) {
    const int set   = wave * 4 + i;       // 0..31
    const int strip = set >> 3;           // 0..3
    const int kc    = set & 7;            // 0..7
    const float* arow = hT + (size_t)(strip * 16 + r15) * DIMz;
    const int kb = kc * 32 + half * 8;
    v16bf hi, lo;
    cvt8(*(const v4f*)(arow + kb),      *(const v4f*)(arow + kb + 4),  0, hi, lo);
    cvt8(*(const v4f*)(arow + kb + 16), *(const v4f*)(arow + kb + 20), 8, hi, lo);
    const size_t off = ((size_t)set * 32 + lane) * 16;
    *(v16bf*)(&smem[off])         = hi;
    *(v16bf*)(&smem[16384 + off]) = lo;
  }
  __syncthreads();

  // ---- Phase B: wave owns N-tiles {2*wid, 2*wid+1} over all 4 strips ----
  const int wid = blockIdx.x * 8 + wave;  // 0..999
#pragma unroll 1
  for (int nn = 0; nn < 2; ++nn) {
    const int nt = wid * 2 + nn;          // 0..1999
    const int N  = nt * 16 + r15;
    v8f acc[4];
    const v8f vz = {0.f, 0.f, 0.f, 0.f, 0.f, 0.f, 0.f, 0.f};
#pragma unroll
    for (int s = 0; s < 4; ++s) acc[s] = vz;

#pragma unroll 1
    for (int kc = 0; kc < 8; ++kc) {
      const size_t bo = ((size_t)(nt * 8 + kc) * 32 + lane) * 16;
      const v16bf Bh = *(const v16bf*)(Hhp + bo);
      const v16bf Bl = *(const v16bf*)(Hlp + bo);
#pragma unroll
      for (int s = 0; s < 4; ++s) {
        const size_t ao = ((size_t)(s * 8 + kc) * 32 + lane) * 16;
        v16bf Ah = *(const v16bf*)(&smem[ao]);
        v16bf Al = *(const v16bf*)(&smem[16384 + ao]);
        acc[s] = WMMA_BF16(Ah, Bh, acc[s]);
        acc[s] = WMMA_BF16(Al, Bh, acc[s]);
        acc[s] = WMMA_BF16(Ah, Bl, acc[s]);
      }
    }
    const float bias = Hb[N];
#pragma unroll
    for (int s = 0; s < 4; ++s) {
#pragma unroll
      for (int rr = 0; rr < 8; ++rr)
        out[(size_t)(s * 16 + half * 8 + rr) * VOCABz + N] = acc[s][rr] + bias;
    }
  }
}

// ---------------------------------------------------------------------------
extern "C" void kernel_launch(void* const* d_in, const int* in_sizes, int n_in,
                              void* d_out, int out_size, void* d_ws, size_t ws_size,
                              hipStream_t stream) {
  const int*   x   = (const int*)d_in[0];
  const float* emb = (const float*)d_in[1];
  const float* Ww  = (const float*)d_in[2];
  const float* Wb  = (const float*)d_in[3];
  const float* Wg  = (const float*)d_in[4];
  const float* Wgb = (const float*)d_in[5];
  const float* Hw  = (const float*)d_in[6];
  const float* Hb  = (const float*)d_in[7];
  float* out = (float*)d_out;

  // Workspace carve-out (256B aligned blocks).
  char* p = (char*)d_ws;
  auto carve = [&](size_t bytes) {
    char* r = p;
    p += (bytes + 255) & ~(size_t)255;
    return r;
  };
  float*  g_buf = (float*)carve((size_t)MTOT * 4);
  float*  S_buf = (float*)carve((size_t)SEG * Bz * DIMz * 4);
  float*  P_buf = (float*)carve((size_t)SEG * Bz * 4);
  float*  h_buf = (float*)carve((size_t)Bz * DIMz * 4);
  __bf16* Whp   = (__bf16*)carve((size_t)8 * 16 * 32 * 16 * 2);           // 128 KB
  __bf16* Wlp   = (__bf16*)carve((size_t)8 * 16 * 32 * 16 * 2);
  __bf16* Hhp   = (__bf16*)carve((size_t)NTILES_HEAD * 8 * 32 * 16 * 2);  // 16.4 MB
  __bf16* Hlp   = (__bf16*)carve((size_t)NTILES_HEAD * 8 * 32 * 16 * 2);
  float*  c_buf = (float*)carve((size_t)MTOT * DIMz * 4);                 // 134 MB

  // Phase 0: pack weights into bf16 hi/lo WMMA fragments (one-time conversions).
  k_pack_w<<<16, 256, 0, stream>>>(Ww, Whp, Wlp);
  k_pack_hw<<<NTILES_HEAD, 256, 0, stream>>>(Hw, Hhp, Hlp);
  // Phase 1: gather + gate + candidate GEMM (LDS-staged A, packed B).
  k_gate_cand<<<MTOT / 128, 256, 131072, stream>>>(x, emb, Whp, Wlp, Wb, Wg, Wgb,
                                                   g_buf, c_buf);
  // Phase 2: segmented scan + combine.
  k_scan_seg<<<Bz * SEG, DIMz, 0, stream>>>(g_buf, c_buf, S_buf, P_buf);
  k_combine<<<Bz, DIMz, 0, stream>>>(S_buf, P_buf, h_buf);
  // Phase 3: head GEMM (LDS-staged A, packed B, head_w read once).
  k_head<<<125, 256, 65536, stream>>>(h_buf, Hhp, Hlp, Hb, out);
}
